// MaskEstimation_37374805410005
// MI455X (gfx1250) — matrix operand order repaired and checked by reading, not compiled, not run on hardware
//
#include <hip/hip_runtime.h>

// ---------------------------------------------------------------------------
// MaskEstimation on gfx1250 (MI455X), f16 WMMA with f32 accumulation.
//
// pass1: z[n][o] = sum_d W1[o,d]*prelu(x[b,d,n]) + b1[o]   (n = k*S+s)
// pass2: out[b,c,o,l] = b2[o] + sum_e W2[o,e] *
//            relu( z[n1][cE+e] + z[n0][cE+e] )             (overlap-add)
// z (64MB per b) lives in workspace, L2-resident (192MB L2) between passes.
// Weights are staged per-block into LDS in exact WMMA A-operand order.
// ---------------------------------------------------------------------------

typedef __attribute__((ext_vector_type(16))) _Float16 v16h;
typedef __attribute__((ext_vector_type(8)))  _Float16 v8h;
typedef __attribute__((ext_vector_type(8)))  float    v8f;
typedef __attribute__((ext_vector_type(4)))  float    v4f;

#define D_DIM   128
#define S_DIM   256
#define KS      65536        // K_DIM * S_DIM
#define O_DIM   256
#define E_DIM   128
#define L_DIM   32768
#define PAD_F   64

// ---------------------------------------------------------------------------
// Weight staging: 256x128 f16 matrix -> LDS, pre-swizzled per WMMA A-operand.
// Slot (tidx = mt*4+kc, lane): plane0 holds A[m][kk+16h .. +8) (16B),
// plane1 holds A[m][kk+16h+8 .. +8) (16B), m = mt*16+(lane&15), h = lane>>4.
// Lane stride 16B -> clean LDS bank pattern for ds_load_b128.
// ---------------------------------------------------------------------------
__device__ __forceinline__ void stage_weights_lds(_Float16* smem,
                                                  const _Float16* __restrict__ wh) {
#pragma unroll
  for (int i = 0; i < 8; ++i) {
    const int slot = threadIdx.x + i * 256;   // 0..2047
    const int lane = slot & 31;
    const int tidx = slot >> 5;               // mt*4 + kc
    const int mt = tidx >> 2, kc = tidx & 3;
    const int m  = mt * 16 + (lane & 15);
    const int h  = (lane >> 4) & 1;
    const int kk = kc * 32;
    const v8h lo = *reinterpret_cast<const v8h*>(wh + m * 128 + kk + 16 * h);
    const v8h hi = *reinterpret_cast<const v8h*>(wh + m * 128 + kk + 16 * h + 8);
    *reinterpret_cast<v8h*>(smem + tidx * 256 + lane * 8)         = lo;
    *reinterpret_cast<v8h*>(smem + 16384 + tidx * 256 + lane * 8) = hi;
  }
  __syncthreads();
}

__device__ __forceinline__ v16h lds_a(const _Float16* smem, int mt, int kc, int lane) {
  const int tidx = mt * 4 + kc;
  const v8h lo = *reinterpret_cast<const v8h*>(smem + tidx * 256 + lane * 8);
  const v8h hi = *reinterpret_cast<const v8h*>(smem + 16384 + tidx * 256 + lane * 8);
  v16h r;
#pragma unroll
  for (int i = 0; i < 8; ++i) { r[i] = lo[i]; r[i + 8] = hi[i]; }
  return r;
}

// One-time convert conv1_w / conv2_w (each 256x128 f32) to f16 in workspace.
__global__ void cvt_weights_kernel(const float* __restrict__ w1,
                                   const float* __restrict__ w2,
                                   _Float16* __restrict__ w1h,
                                   _Float16* __restrict__ w2h) {
  const int i = blockIdx.x * blockDim.x + threadIdx.x;   // 0..65535
  if (i < 32768) w1h[i] = (_Float16)w1[i];
  else           w2h[i - 32768] = (_Float16)w2[i - 32768];
}

// ---------------------------------------------------------------------------
// pass1: per wave, output strip M=256 (all o) x N=16 (n = k*S + s, s fastest
// -> coalesced x reads).  K=128 in 4 chunks of 32.  z stored as z[n][o].
// ---------------------------------------------------------------------------
__global__ __launch_bounds__(256) void pass1_kernel(
    const float* __restrict__ x, const float* __restrict__ prelu_w,
    const _Float16* __restrict__ w1h, const float* __restrict__ b1,
    float* __restrict__ z, int b) {
  __shared__ __align__(16) _Float16 smem[32768];   // 64KB
  stage_weights_lds(smem, w1h);

  const int lane = threadIdx.x & 31;
  const int wid  = blockIdx.x * 8 + (threadIdx.x >> 5);   // 0..4095
  const int h    = (lane >> 4) & 1;
  const int n    = wid * 16 + (lane & 15);                // flat (k,s) column
  const float pw = prelu_w[0];
  const float* __restrict__ xb = x + (size_t)b * D_DIM * KS + n;

  v8f acc[16];
#pragma unroll
  for (int mt = 0; mt < 16; ++mt)
#pragma unroll
    for (int r = 0; r < 8; ++r) acc[mt][r] = b1[mt * 16 + 8 * h + r];

#pragma unroll
  for (int kc = 0; kc < 4; ++kc) {
    const int kk = kc * 32;
    // B-operand (32x16 K x N): lane half h holds K = kk+16h .. +15 of col n.
    v16h bm;
#pragma unroll
    for (int j = 0; j < 16; ++j) {
      const int d = kk + 16 * h + j;
      float xv = __builtin_nontemporal_load(xb + (size_t)d * KS);
      xv = (xv >= 0.f) ? xv : pw * xv;
      bm[j] = (_Float16)xv;
    }
#pragma unroll
    for (int mt = 0; mt < 16; ++mt) {
      const v16h am = lds_a(smem, mt, kc, lane);
      acc[mt] = __builtin_amdgcn_wmma_f32_16x16x32_f16(
          false, am, false, bm, (short)0, acc[mt], false, false);
    }
  }

  // z[n][o]: per lane a contiguous 32B chunk per mt; lanes i and 16+i share n
  // and cover adjacent 32B halves -> full 64B blocks.
  float* __restrict__ zn = z + (size_t)n * O_DIM;
#pragma unroll
  for (int mt = 0; mt < 16; ++mt)
    *reinterpret_cast<v8f*>(zn + mt * 16 + 8 * h) = acc[mt];
}

// ---------------------------------------------------------------------------
// pass2: overlap-add + ReLU fused into the B-operand, then GEMM2.
// Per wave: M=256 (all o2) x N=16 (l-tile), K=128 (e) in 4 chunks of 32.
// z[n][o] layout -> per-lane contiguous 64B reads; lanes i / 16+i share l and
// read adjacent e-halves -> full 128B lines from L2-resident z.
// ---------------------------------------------------------------------------
__global__ __launch_bounds__(256) void pass2_kernel(
    const float* __restrict__ z, const _Float16* __restrict__ w2h,
    const float* __restrict__ b2, float* __restrict__ out, int b) {
  __shared__ __align__(16) _Float16 smem[32768];   // 64KB
  stage_weights_lds(smem, w2h);

  const int lane = threadIdx.x & 31;
  const int wid  = blockIdx.x * 8 + (threadIdx.x >> 5);   // 0..4095
  const int c    = wid >> 11;                             // 0..1
  const int lt   = wid & 2047;                            // 16-wide l-tile
  const int h    = (lane >> 4) & 1;
  const int l    = lt * 16 + (lane & 15);
  const int t    = l + PAD_F;        // position in un-sliced OLA output
  const int s1   = t >> 7;           // frame with k in [0,128)
  const int k1   = t & 127;
  const bool hi_ok = (s1 < S_DIM);
  const bool lo_ok = (s1 > 0);
  const int s1c = hi_ok ? s1 : (S_DIM - 1);
  const int s0c = lo_ok ? (s1 - 1) : 0;
  // Base pointers for the two overlapping frames (always in-bounds; masked).
  const float* __restrict__ z1 = z + (size_t)(k1 * S_DIM + s1c) * O_DIM + c * E_DIM;
  const float* __restrict__ z0 = z + (size_t)((k1 + 128) * S_DIM + s0c) * O_DIM + c * E_DIM;

  v8f acc[16];
#pragma unroll
  for (int mt = 0; mt < 16; ++mt)
#pragma unroll
    for (int r = 0; r < 8; ++r) acc[mt][r] = b2[mt * 16 + 8 * h + r];

#pragma unroll
  for (int kc = 0; kc < 4; ++kc) {
    const int kk = kc * 32;
    const int eo = kk + 16 * h;      // e-offset of this lane half
    v4f va[4], vb[4];
#pragma unroll
    for (int q = 0; q < 4; ++q) {
      va[q] = *reinterpret_cast<const v4f*>(z1 + eo + q * 4);
      vb[q] = *reinterpret_cast<const v4f*>(z0 + eo + q * 4);
    }
    v16h bm;
#pragma unroll
    for (int j = 0; j < 16; ++j) {
      float v = (hi_ok ? va[j >> 2][j & 3] : 0.f) +
                (lo_ok ? vb[j >> 2][j & 3] : 0.f);
      v = (v > 0.f) ? v : 0.f;                         // ReLU
      bm[j] = (_Float16)v;
    }
#pragma unroll
    for (int mt = 0; mt < 16; ++mt) {
      const v16h am = lds_a(smem, mt, kc, lane);
      acc[mt] = __builtin_amdgcn_wmma_f32_16x16x32_f16(
          false, am, false, bm, (short)0, acc[mt], false, false);
    }
  }

  float* __restrict__ ob = out + (size_t)(b * 2 + c) * O_DIM * L_DIM + l;
#pragma unroll
  for (int mt = 0; mt < 16; ++mt)
#pragma unroll
    for (int r = 0; r < 8; ++r) {
      const int o = mt * 16 + 8 * h + r;
      __builtin_nontemporal_store(acc[mt][r], ob + (size_t)o * L_DIM);
    }
}

extern "C" void kernel_launch(void* const* d_in, const int* in_sizes, int n_in,
                              void* d_out, int out_size, void* d_ws, size_t ws_size,
                              hipStream_t stream) {
  const float* x  = (const float*)d_in[0];
  const float* pw = (const float*)d_in[1];
  const float* w1 = (const float*)d_in[2];
  const float* b1 = (const float*)d_in[3];
  const float* w2 = (const float*)d_in[4];
  const float* b2 = (const float*)d_in[5];
  float* out = (float*)d_out;

  char* ws = (char*)d_ws;
  _Float16* w1h = (_Float16*)(ws);            // 64 KB
  _Float16* w2h = (_Float16*)(ws + 65536);    // 64 KB
  float*    z   = (float*)(ws + 131072);      // 64 MB, reused per batch

  cvt_weights_kernel<<<256, 256, 0, stream>>>(w1, w2, w1h, w2h);
  for (int b = 0; b < 4; ++b) {
    // Stream order serializes pass1(b) -> pass2(b) -> pass1(b+1): the 64MB z
    // region is safely reused and stays L2-resident between the two passes.
    pass1_kernel<<<512, 256, 0, stream>>>(x, pw, w1h, b1, z, b);
    pass2_kernel<<<512, 256, 0, stream>>>(z, w2h, b2, out, b);
  }
}
